// Attention_2052994367753
// MI455X (gfx1250) — matrix operand order repaired and checked by reading, not compiled, not run on hardware
//
#include <hip/hip_runtime.h>

#define SEQ    2048
#define BATCH  32
#define KSPLIT 4
#define KCHUNK (SEQ / KSPLIT)   // 512

typedef __attribute__((ext_vector_type(16))) __bf16 v16bf;
typedef __attribute__((ext_vector_type(8)))  float  v8f;

// ---------------------------------------------------------------------------
// Kernel 1: fused QKV GEMM, K-split for bandwidth saturation.
// partial[ks][z] = x[:, ksplit] @ W[z][:, ksplit]^T      (no bias yet)
// M=32, N=2048, K=512 per block.  bf16 WMMA 16x16x32, f32 accumulate.
// Block = 256 threads = 8 waves; each wave owns a 32(M) x 16(N) tile
// (two 16x16 accumulators sharing each B tile -> every W element read once).
// grid = (SEQ/128, KSPLIT, 3).  1536 waves total -> enough loads in flight
// to approach 23.3 TB/s on the 50 MB weight stream.
// ---------------------------------------------------------------------------
__global__ __launch_bounds__(256) void qkv_gemm_wmma(
    const float* __restrict__ x,
    const float* __restrict__ Wq,
    const float* __restrict__ Wk,
    const float* __restrict__ Wv,
    float* __restrict__ partial)   // [KSPLIT][3][BATCH][SEQ]
{
    const int z    = blockIdx.z;
    const int ks   = blockIdx.y;
    const float* W = (z == 0) ? Wq : (z == 1) ? Wk : Wv;
    float* out     = partial + ((size_t)ks * 3 + z) * BATCH * SEQ;

    const int lane = threadIdx.x & 31;
    const int wave = threadIdx.x >> 5;
    const int n0   = blockIdx.x * 128 + wave * 16;   // wave's N base
    const int k0   = ks * KCHUNK;

    // A layout (16-bit A 16x32): lane m = lane&15; lanes 0-15 hold K {0-7,16-23},
    // lanes 16-31 hold K {8-15,24-31}; VGPR j holds pair (k0(j), k0(j)+1).
    const int m  = lane & 15;
    const int kb = (lane >> 4) ? 8 : 0;

    v8f c0 = {};   // rows M = 0..15
    v8f c1 = {};   // rows M = 16..31

#pragma unroll 2
    for (int kk = k0; kk < k0 + KCHUNK; kk += 32) {
        v16bf a0, a1, bt;
#pragma unroll
        for (int vg = 0; vg < 8; ++vg) {
            const int k = ((vg < 4) ? (vg * 2) : (16 + (vg - 4) * 2)) + kb;
            a0[2 * vg + 0] = (__bf16)x[m * SEQ + kk + k];
            a0[2 * vg + 1] = (__bf16)x[m * SEQ + kk + k + 1];
            a1[2 * vg + 0] = (__bf16)x[(m + 16) * SEQ + kk + k];
            a1[2 * vg + 1] = (__bf16)x[(m + 16) * SEQ + kk + k + 1];
            // B = W^T : B(k,n) = W[n][k]; lane holds column n = lane&15,
            // elements use the same K order as A.
            bt[2 * vg + 0] = (__bf16)W[(size_t)(n0 + m) * SEQ + kk + k];
            bt[2 * vg + 1] = (__bf16)W[(size_t)(n0 + m) * SEQ + kk + k + 1];
        }
        c0 = __builtin_amdgcn_wmma_f32_16x16x32_bf16(false, a0, false, bt,
                                                     (short)0, c0, false, false);
        c1 = __builtin_amdgcn_wmma_f32_16x16x32_bf16(false, a1, false, bt,
                                                     (short)0, c1, false, false);
    }

    // C/D layout: lanes 0-15 -> N = lane, M = vgpr; lanes 16-31 -> N = lane-16, M = vgpr+8.
    const int n = n0 + (lane & 15);
#pragma unroll
    for (int vg = 0; vg < 8; ++vg) {
        const int mr = vg + ((lane >> 4) ? 8 : 0);
        out[(size_t)mr        * SEQ + n] = c0[vg];
        out[(size_t)(mr + 16) * SEQ + n] = c1[vg];
    }
}

// ---------------------------------------------------------------------------
// Kernel 1b: reduce K-splits + bias.  qkv[z][b][n] = sum_ks partial + bias[z][n]
// grid = (3*BATCH*SEQ/256), block = 256.  Deterministic (no FP atomics).
// ---------------------------------------------------------------------------
__global__ __launch_bounds__(256) void qkv_reduce(
    const float* __restrict__ partial,
    const float* __restrict__ bq, const float* __restrict__ bk,
    const float* __restrict__ bv,
    float* __restrict__ qkv)       // [3][BATCH][SEQ]
{
    const int t = blockIdx.x * 256 + threadIdx.x;   // 0 .. 3*BATCH*SEQ-1
    const int z = t / (BATCH * SEQ);
    const int r = t - z * (BATCH * SEQ);
    const int n = r & (SEQ - 1);
    const float* B = (z == 0) ? bq : (z == 1) ? bk : bv;

    float s = B[n];
#pragma unroll
    for (int ks = 0; ks < KSPLIT; ++ks)
        s += partial[((size_t)ks * 3 + z) * BATCH * SEQ + r];
    qkv[t] = s;
}

// ---------------------------------------------------------------------------
// Kernel 2: per-row softmax denominators.  w[b][i] = v[b][i] / sum_j exp(q_i*k_j)
// grid = (SEQ/256, BATCH), block = 256.  k[b][:] staged in 8 KB LDS.
// ---------------------------------------------------------------------------
__global__ __launch_bounds__(256) void attn_pass1(
    const float* __restrict__ qkv, float* __restrict__ w)
{
    __shared__ float ks[SEQ];
    const int b = blockIdx.y;
    const float* qp = qkv + 0 * BATCH * SEQ + (size_t)b * SEQ;
    const float* kp = qkv + 1 * BATCH * SEQ + (size_t)b * SEQ;
    const float* vp = qkv + 2 * BATCH * SEQ + (size_t)b * SEQ;

    for (int t = threadIdx.x; t < SEQ; t += 256) ks[t] = kp[t];
    __syncthreads();

    const int   i  = blockIdx.x * 256 + threadIdx.x;
    const float qi = qp[i];
    const float4* k4 = (const float4*)ks;
    float Z = 0.f;
#pragma unroll 4
    for (int j = 0; j < SEQ / 4; ++j) {
        const float4 kv = k4[j];
        Z += __expf(qi * kv.x) + __expf(qi * kv.y) +
             __expf(qi * kv.z) + __expf(qi * kv.w);
    }
    w[(size_t)b * SEQ + i] = vp[i] / Z;
}

// ---------------------------------------------------------------------------
// Kernel 3: sa[b][j] = sum_i w[b][i] * exp(q[b][i]*k[b][j]);  out = sa + x.
// grid = (SEQ/256, BATCH), block = 256.  q and w staged in 16 KB LDS.
// ---------------------------------------------------------------------------
__global__ __launch_bounds__(256) void attn_pass2(
    const float* __restrict__ qkv, const float* __restrict__ w,
    const float* __restrict__ x, float* __restrict__ out)
{
    __shared__ float qs[SEQ];
    __shared__ float wsm[SEQ];
    const int b = blockIdx.y;
    const float* qp = qkv + 0 * BATCH * SEQ + (size_t)b * SEQ;
    const float* kp = qkv + 1 * BATCH * SEQ + (size_t)b * SEQ;
    const float* wp = w + (size_t)b * SEQ;

    for (int t = threadIdx.x; t < SEQ; t += 256) { qs[t] = qp[t]; wsm[t] = wp[t]; }
    __syncthreads();

    const int   j  = blockIdx.x * 256 + threadIdx.x;
    const float kj = kp[j];
    const float4* q4 = (const float4*)qs;
    const float4* w4 = (const float4*)wsm;
    float acc = 0.f;
#pragma unroll 4
    for (int i = 0; i < SEQ / 4; ++i) {
        const float4 qv = q4[i];
        const float4 wv = w4[i];
        acc += wv.x * __expf(qv.x * kj) + wv.y * __expf(qv.y * kj) +
               wv.z * __expf(qv.z * kj) + wv.w * __expf(qv.w * kj);
    }
    out[(size_t)b * SEQ + j] = acc + x[(size_t)b * SEQ + j];
}

// ---------------------------------------------------------------------------
extern "C" void kernel_launch(void* const* d_in, const int* in_sizes, int n_in,
                              void* d_out, int out_size, void* d_ws, size_t ws_size,
                              hipStream_t stream) {
    const float* x  = (const float*)d_in[0];
    const float* Wq = (const float*)d_in[1];
    const float* bq = (const float*)d_in[2];
    const float* Wk = (const float*)d_in[3];
    const float* bk = (const float*)d_in[4];
    const float* Wv = (const float*)d_in[5];
    const float* bv = (const float*)d_in[6];
    float* out = (float*)d_out;

    float* partial = (float*)d_ws;                        // 4*3*32*2048 f32 = 3 MiB
    float* qkv     = partial + KSPLIT * 3 * BATCH * SEQ;  // 3*32*2048 f32 = 768 KiB
    float* w       = qkv + 3 * BATCH * SEQ;               // 32*2048 f32   = 256 KiB

    qkv_gemm_wmma<<<dim3(SEQ / 128, KSPLIT, 3), 256, 0, stream>>>(
        x, Wq, Wk, Wv, partial);
    qkv_reduce<<<dim3(3 * BATCH * SEQ / 256), 256, 0, stream>>>(
        partial, bq, bk, bv, qkv);
    attn_pass1<<<dim3(SEQ / 256, BATCH), 256, 0, stream>>>(qkv, w);
    attn_pass2<<<dim3(SEQ / 256, BATCH), 256, 0, stream>>>(qkv, w, x, out);
}
